// EventGAT_5686536700291
// MI455X (gfx1250) — matrix operand order repaired and checked by reading, not compiled, not run on hardware
//
#include <hip/hip_runtime.h>

// ---------------------------------------------------------------------------
// GAT (2-layer, PyG GATConv semantics) for MI455X / gfx1250, wave32 + WMMA.
// N=100000 nodes, E=1.6M edges (+N self loops), IN=HID=128, HEADS=4, OUT=64.
// ---------------------------------------------------------------------------

typedef __attribute__((ext_vector_type(16))) __bf16 v16bf;
typedef __attribute__((ext_vector_type(8)))  __bf16 v8bf;
typedef __attribute__((ext_vector_type(8)))  float  v8f;

#define NEG_SLOPE 0.2f

// ---------------------------------------------------------------------------
// Fill kernel (also used for zeroing accumulators / -inf for segment max).
// ---------------------------------------------------------------------------
__global__ __launch_bounds__(256) void gat_fill_f32(float* __restrict__ p, float v, int n) {
    int i = blockIdx.x * blockDim.x + threadIdx.x;
    if (i < n) p[i] = v;
}

// ---------------------------------------------------------------------------
// WMMA GEMM: H[M x NCOLS] = X[M x 128] * W[128 x NCOLS]
// bf16 operands (converted on the fly), f32 accumulate.
// Block: 256 threads = 8 waves; tile 128 rows x NCOLS cols; K in steps of 32.
//
// Fragment layouts used here (wave32, contiguous k order for A and B):
//   A (16x32 bf16): lane half (lane>>4) selects K-offset pattern; for lane m,
//     afrag element j holds K = half*16 + j.
//     -> two contiguous 8-element (16B) LDS reads per lane.
//   B (32x16 bf16): lanes 0-15 column N=lane hold K=0..15 (pairs per VGPR),
//     lanes 16-31 hold K=16..31 -> 16 contiguous bf16 per lane when W is
//     stored K-major (transposed) in LDS: one 32B read.
//   C/D (16x16 f32): VGPR r: lanes 0-15 -> (M=r, N=lane); lanes 16-31 ->
//     (M=r+8, N=lane-16).
// ---------------------------------------------------------------------------
template <int NCOLS>
__global__ __launch_bounds__(256) void gat_gemm_wmma(const float* __restrict__ X,
                                                     const float* __restrict__ W,
                                                     float* __restrict__ H, int M) {
    constexpr int NT = NCOLS / 16;          // 16x16 output tiles per wave
    __shared__ __bf16 lWt[NCOLS][144];      // W transposed: [n][k], 288B rows (32B aligned)
    __shared__ __bf16 lX[128][40];          // X tile: [m][k], 80B rows (16B aligned)

    const int tid  = threadIdx.x;
    const int wave = tid >> 5;
    const int lane = tid & 31;
    const int half = lane >> 4;
    const int l16  = lane & 15;
    const int rowBase = blockIdx.x * 128;

    // Stage W (128 x NCOLS) into LDS, transposed, converted to bf16.
    for (int i = tid; i < NCOLS * 128; i += 256) {
        int n = i % NCOLS;                  // coalesced global read
        int k = i / NCOLS;
        lWt[n][k] = (__bf16)W[(size_t)k * NCOLS + n];
    }

    v8f acc[NT];
#pragma unroll
    for (int t = 0; t < NT; ++t)
        acc[t] = (v8f){0.f, 0.f, 0.f, 0.f, 0.f, 0.f, 0.f, 0.f};

    for (int k0 = 0; k0 < 128; k0 += 32) {
        __syncthreads();                    // lX reuse guard (+W ready on iter 0)
        // Stage 128x32 slice of X, converted to bf16 (coalesced 128B rows).
        for (int i = tid; i < 128 * 32; i += 256) {
            int m = i >> 5, k = i & 31;
            int row = rowBase + m;
            float v = (row < M) ? X[(size_t)row * 128 + k0 + k] : 0.f;
            lX[m][k] = (__bf16)v;
        }
        __syncthreads();

        // A fragment for this wave's 16 rows.
        const int mrow = wave * 16 + l16;
        v8bf alo = *(const v8bf*)&lX[mrow][half * 16];
        v8bf ahi = *(const v8bf*)&lX[mrow][half * 16 + 8];
        v16bf afrag = __builtin_shufflevector(alo, ahi, 0, 1, 2, 3, 4, 5, 6, 7,
                                              8, 9, 10, 11, 12, 13, 14, 15);
#pragma unroll
        for (int t = 0; t < NT; ++t) {
            v16bf bfrag = *(const v16bf*)&lWt[t * 16 + l16][k0 + half * 16];
            acc[t] = __builtin_amdgcn_wmma_f32_16x16x32_bf16(
                /*neg_a=*/false, afrag, /*neg_b=*/false, bfrag,
                /*c_mod=*/(short)0, acc[t], /*reuse_a=*/false, /*reuse_b=*/false);
        }
    }

    // Store D tiles (f32).
    const int mOut = rowBase + wave * 16 + half * 8;
#pragma unroll
    for (int t = 0; t < NT; ++t) {
#pragma unroll
        for (int r = 0; r < 8; ++r) {
            int row = mOut + r;
            if (row < M) H[(size_t)row * NCOLS + t * 16 + l16] = acc[t][r];
        }
    }
}

// ---------------------------------------------------------------------------
// Attention logits: al_s[n,h] = <H[n, h*ch : (h+1)*ch], a_src[h]>, same for dst.
// ---------------------------------------------------------------------------
__global__ __launch_bounds__(256) void gat_att_logits(const float* __restrict__ H,
                                                      const float* __restrict__ asrc,
                                                      const float* __restrict__ adst,
                                                      float* __restrict__ als,
                                                      float* __restrict__ ald,
                                                      int Nn, int heads, int ch) {
    int idx = blockIdx.x * blockDim.x + threadIdx.x;
    if (idx >= Nn * heads) return;
    int node = idx / heads, hd = idx % heads;
    const float* row = H + (size_t)node * heads * ch + hd * ch;
    float ss = 0.f, sd = 0.f;
    for (int c = 0; c < ch; ++c) {
        float v = row[c];
        ss += v * asrc[hd * ch + c];
        sd += v * adst[hd * ch + c];
    }
    als[idx] = ss;
    ald[idx] = sd;
}

__device__ __forceinline__ float gat_lrelu(float v) {
    return (v > 0.f) ? v : NEG_SLOPE * v;
}

// ---------------------------------------------------------------------------
// Pass 1 over edges: segment max of leaky-relu logits per (dst, head).
// Edges e < E come from edge_index; e >= E are self loops (s = d = e - E).
// ---------------------------------------------------------------------------
template <int HEADS_>
__global__ __launch_bounds__(256) void gat_edge_max(const int* __restrict__ ei,
                                                    const float* __restrict__ als,
                                                    const float* __restrict__ ald,
                                                    float* __restrict__ emax,
                                                    int E, int Nn) {
    int e = blockIdx.x * blockDim.x + threadIdx.x;
    if (e >= E + Nn) return;
    int s = (e < E) ? ei[e]     : (e - E);
    int d = (e < E) ? ei[E + e] : (e - E);
#pragma unroll
    for (int h = 0; h < HEADS_; ++h) {
        float v = gat_lrelu(als[s * HEADS_ + h] + ald[d * HEADS_ + h]);
        __hip_atomic_fetch_max(&emax[d * HEADS_ + h], v,
                               __ATOMIC_RELAXED, __HIP_MEMORY_SCOPE_AGENT);
    }
}

// ---------------------------------------------------------------------------
// Pass 2 over edges (fused): acc[d,:] += exp(logit - max) * H[s,:]
//                            esum[d,h] += exp(logit - max)
// Softmax division is deferred to the finalize kernel. One wave per edge;
// 32 lanes stride the C channels -> fully coalesced 128B gathers + f32 atomics.
// ---------------------------------------------------------------------------
template <int C, int HEADS_>
__global__ __launch_bounds__(256) void gat_edge_aggr(const int* __restrict__ ei,
                                                     const float* __restrict__ H,
                                                     const float* __restrict__ als,
                                                     const float* __restrict__ ald,
                                                     const float* __restrict__ emax,
                                                     float* __restrict__ acc,
                                                     float* __restrict__ esum,
                                                     int E, int Nn) {
    int wid  = blockIdx.x * 8 + (threadIdx.x >> 5);
    int lane = threadIdx.x & 31;
    if (wid >= E + Nn) return;
    int s = (wid < E) ? ei[wid]     : (wid - E);
    int d = (wid < E) ? ei[E + wid] : (wid - E);
    constexpr int CH = C / HEADS_;          // channels per head
#pragma unroll
    for (int i = 0; i < C / 32; ++i) {
        const int hd = (i * 32) / CH;       // head is uniform across the wave
        float v  = gat_lrelu(als[s * HEADS_ + hd] + ald[d * HEADS_ + hd]);
        float ex = __expf(v - emax[d * HEADS_ + hd]);
        int c = i * 32 + lane;
        atomicAdd(&acc[(size_t)d * C + c], ex * H[(size_t)s * C + c]);
        if (lane == 0 && (i * 32) % CH == 0)  // once per (edge, head)
            atomicAdd(&esum[d * HEADS_ + hd], ex);
    }
}

// ---------------------------------------------------------------------------
// Finalize: out = acc / esum[head] + bias  (optional ReLU). In-place safe.
// ---------------------------------------------------------------------------
template <int C, int HEADS_>
__global__ __launch_bounds__(256) void gat_finalize(const float* __restrict__ acc,
                                                    const float* __restrict__ esum,
                                                    const float* __restrict__ bias,
                                                    float* __restrict__ out,
                                                    int Nn, int do_relu) {
    int i = blockIdx.x * blockDim.x + threadIdx.x;
    if (i >= Nn * C) return;
    int node = i / C, c = i % C;
    float den = esum[node * HEADS_ + c / (C / HEADS_)];
    float v = acc[i] / den + bias[c];
    if (do_relu) v = fmaxf(v, 0.f);
    out[i] = v;
}

// ---------------------------------------------------------------------------
// Host-side orchestration.
// ---------------------------------------------------------------------------
extern "C" void kernel_launch(void* const* d_in, const int* in_sizes, int n_in,
                              void* d_out, int out_size, void* d_ws, size_t ws_size,
                              hipStream_t stream) {
    (void)n_in; (void)out_size; (void)ws_size;

    const float* x   = (const float*)d_in[0];
    const int*   ei  = (const int*)  d_in[1];   // [2, E] row-major (src row, dst row)
    const float* W1  = (const float*)d_in[2];   // [128, 128]
    const float* as1 = (const float*)d_in[3];   // [4, 32]
    const float* ad1 = (const float*)d_in[4];
    const float* b1  = (const float*)d_in[5];   // [128]
    const float* W2  = (const float*)d_in[6];   // [128, 64]
    const float* as2 = (const float*)d_in[7];   // [1, 64]
    const float* ad2 = (const float*)d_in[8];
    const float* b2  = (const float*)d_in[9];   // [64]

    const int N = in_sizes[0] / 128;
    const int E = in_sizes[1] / 2;
    const int TOT = E + N;                      // edges incl. self loops

    // --- workspace carve-up (with aliasing; ~136 MB total) ---
    char* base = (char*)d_ws;
    size_t off = 0;
    auto carve = [&](size_t bytes) -> float* {
        float* p = (float*)(base + off);
        off += (bytes + 255) & ~(size_t)255;
        return p;
    };
    float* h1    = carve((size_t)N * 128 * 4);  // layer-1 features (aliased as acc2 later)
    float* acc1  = carve((size_t)N * 128 * 4);  // layer-1 accumulator -> x2 after finalize
    float* h2    = carve((size_t)N * 64 * 4);   // layer-2 features
    float* als1  = carve((size_t)N * 4 * 4);
    float* ald1  = carve((size_t)N * 4 * 4);
    float* emax1 = carve((size_t)N * 4 * 4);
    float* esum1 = carve((size_t)N * 4 * 4);
    float* als2  = carve((size_t)N * 4);
    float* ald2  = carve((size_t)N * 4);
    float* emax2 = carve((size_t)N * 4);
    float* esum2 = carve((size_t)N * 4);
    float* acc2  = h1;                          // h1 dead after layer-1 aggregation

    const int TPB = 256;
    auto blocks = [](long long n, int tpb) { return (unsigned)((n + tpb - 1) / tpb); };

    // ================= Layer 1 =================
    gat_fill_f32<<<blocks((long long)N * 4, TPB), TPB, 0, stream>>>(emax1, -1e30f, N * 4);
    gat_fill_f32<<<blocks((long long)N * 4, TPB), TPB, 0, stream>>>(esum1, 0.f, N * 4);
    gat_fill_f32<<<blocks((long long)N * 128, TPB), TPB, 0, stream>>>(acc1, 0.f, N * 128);

    gat_gemm_wmma<128><<<blocks(N, 128), TPB, 0, stream>>>(x, W1, h1, N);
    gat_att_logits<<<blocks((long long)N * 4, TPB), TPB, 0, stream>>>(h1, as1, ad1,
                                                                      als1, ald1, N, 4, 32);
    gat_edge_max<4><<<blocks(TOT, TPB), TPB, 0, stream>>>(ei, als1, ald1, emax1, E, N);
    gat_edge_aggr<128, 4><<<blocks(TOT, 8), TPB, 0, stream>>>(ei, h1, als1, ald1,
                                                              emax1, acc1, esum1, E, N);
    gat_finalize<128, 4><<<blocks((long long)N * 128, TPB), TPB, 0, stream>>>(
        acc1, esum1, b1, acc1, N, /*relu=*/1);   // acc1 now holds x2

    // ================= Layer 2 =================
    gat_fill_f32<<<blocks(N, TPB), TPB, 0, stream>>>(emax2, -1e30f, N);
    gat_fill_f32<<<blocks(N, TPB), TPB, 0, stream>>>(esum2, 0.f, N);
    gat_fill_f32<<<blocks((long long)N * 64, TPB), TPB, 0, stream>>>(acc2, 0.f, N * 64);

    gat_gemm_wmma<64><<<blocks(N, 128), TPB, 0, stream>>>(acc1, W2, h2, N);
    gat_att_logits<<<blocks(N, TPB), TPB, 0, stream>>>(h2, as2, ad2, als2, ald2, N, 1, 64);
    gat_edge_max<1><<<blocks(TOT, TPB), TPB, 0, stream>>>(ei, als2, ald2, emax2, E, N);
    gat_edge_aggr<64, 1><<<blocks(TOT, 8), TPB, 0, stream>>>(ei, h2, als2, ald2,
                                                             emax2, acc2, esum2, E, N);
    gat_finalize<64, 1><<<blocks((long long)N * 64, TPB), TPB, 0, stream>>>(
        acc2, esum2, b2, (float*)d_out, N, /*relu=*/0);
}
